// RingLlamaAttention_17128329576642
// MI455X (gfx1250) — matrix operand order repaired and checked by reading, not compiled, not run on hardware
//
#include <hip/hip_runtime.h>
#include <hip/hip_bf16.h>

// ---------------------------------------------------------------------------
// Problem constants (fixed by setup_inputs in the reference)
// ---------------------------------------------------------------------------
#define BATCH    2
#define SEQ      4096
#define HID      4096
#define WORLD    4
#define SLOCAL   (SEQ / WORLD)     // 1024
#define NH       32
#define HD       128

typedef _Float16 v16h __attribute__((ext_vector_type(16)));
typedef _Float16 v8h  __attribute__((ext_vector_type(8)));
typedef _Float16 v4h  __attribute__((ext_vector_type(4)));
typedef float    v8f  __attribute__((ext_vector_type(8)));

#define WMMA_F16(a, b, c) \
  __builtin_amdgcn_wmma_f32_16x16x32_f16(false, (a), false, (b), (short)0, (c), false, false)

static __device__ __forceinline__ v16h cat8(v8h lo, v8h hi) {
  return __builtin_shufflevector(lo, hi, 0, 1, 2, 3, 4, 5, 6, 7,
                                         8, 9, 10, 11, 12, 13, 14, 15);
}

// A-fragment (16x32 f16, M x K): lane holds row m = lane%16.
// halves 0..15 = A[m][k0 + g*16 .. +16] (contiguous K, matches B-fragment)
static __device__ __forceinline__ v16h load_a_frag(const _Float16* __restrict__ rowbase,
                                                   long long lda, int m, int k0,
                                                   int g) {
  const _Float16* p = rowbase + (long long)m * lda + k0 + g * 16;
  v8h lo = *reinterpret_cast<const v8h*>(p);
  v8h hi = *reinterpret_cast<const v8h*>(p + 8);
  return cat8(lo, hi);
}

// B-fragment (32x16 f16, K x N): lane<16 -> N=lane, K=0..15 contiguous;
// lane>=16 -> N=lane-16, K=16..31 contiguous. One 32B load per lane.
static __device__ __forceinline__ v16h load_b_frag(const _Float16* __restrict__ colbase,
                                                   long long ldb, int n, int k0,
                                                   int g) {
  const _Float16* p = colbase + (long long)n * ldb + k0 + g * 16;
  return *reinterpret_cast<const v16h*>(p);
}

static __device__ __forceinline__ float redmax16(float v) {
  v = fmaxf(v, __shfl_xor(v, 1, 32));
  v = fmaxf(v, __shfl_xor(v, 2, 32));
  v = fmaxf(v, __shfl_xor(v, 4, 32));
  v = fmaxf(v, __shfl_xor(v, 8, 32));
  return v;
}
static __device__ __forceinline__ float redsum16(float v) {
  v += __shfl_xor(v, 1, 32);
  v += __shfl_xor(v, 2, 32);
  v += __shfl_xor(v, 4, 32);
  v += __shfl_xor(v, 8, 32);
  return v;
}

// ---------------------------------------------------------------------------
// fp32 -> f16 conversion (vectorized x4)
// ---------------------------------------------------------------------------
__global__ __launch_bounds__(256) void cvt_f32_f16(const float* __restrict__ src,
                                                   _Float16* __restrict__ dst,
                                                   long long n4) {
  long long i = (long long)blockIdx.x * blockDim.x + threadIdx.x;
  if (i < n4) {
    float4 v = reinterpret_cast<const float4*>(src)[i];
    v4h h = {(_Float16)v.x, (_Float16)v.y, (_Float16)v.z, (_Float16)v.w};
    reinterpret_cast<v4h*>(dst)[i] = h;
  }
}

// ---------------------------------------------------------------------------
// Tiled WMMA GEMM:  C[m,n] = sum_k A[m,k] * W[n,k]
// Block tile 128x128 (8 waves: 4 along M x 2 along N, each wave 32x64)
// ---------------------------------------------------------------------------
__global__ __launch_bounds__(256) void gemm_wmma(const _Float16* __restrict__ A,
                                                 const _Float16* __restrict__ W,
                                                 void* __restrict__ Cout,
                                                 int K, long long lda, long long ldb,
                                                 long long ldc,
                                                 long long a_bstride, long long c_bstride,
                                                 int out_f16) {
  const int lane = threadIdx.x & 31;
  const int wave = threadIdx.x >> 5;
  const int ln = lane & 15;
  const int g  = lane >> 4;
  const int wm = (wave & 3) * 32;   // wave row offset within block tile
  const int wn = (wave >> 2) * 64;  // wave col offset within block tile
  const int m_blk = blockIdx.y * 128;
  const int n_blk = blockIdx.x * 128;

  const _Float16* Ab = A + (long long)blockIdx.z * a_bstride
                         + (long long)(m_blk + wm) * lda;
  const _Float16* Wb = W + (long long)(n_blk + wn) * ldb;

  v8f zero = {0.f, 0.f, 0.f, 0.f, 0.f, 0.f, 0.f, 0.f};
  v8f acc[2][4];
#pragma unroll
  for (int i = 0; i < 2; ++i)
#pragma unroll
    for (int j = 0; j < 4; ++j) acc[i][j] = zero;

  for (int k0 = 0; k0 < K; k0 += 32) {
    v16h a0 = load_a_frag(Ab, lda, ln,      k0, g);
    v16h a1 = load_a_frag(Ab, lda, 16 + ln, k0, g);
    v16h b0 = load_b_frag(Wb, ldb, ln,      k0, g);
    v16h b1 = load_b_frag(Wb, ldb, 16 + ln, k0, g);
    v16h b2 = load_b_frag(Wb, ldb, 32 + ln, k0, g);
    v16h b3 = load_b_frag(Wb, ldb, 48 + ln, k0, g);
    acc[0][0] = WMMA_F16(a0, b0, acc[0][0]);
    acc[0][1] = WMMA_F16(a0, b1, acc[0][1]);
    acc[0][2] = WMMA_F16(a0, b2, acc[0][2]);
    acc[0][3] = WMMA_F16(a0, b3, acc[0][3]);
    acc[1][0] = WMMA_F16(a1, b0, acc[1][0]);
    acc[1][1] = WMMA_F16(a1, b1, acc[1][1]);
    acc[1][2] = WMMA_F16(a1, b2, acc[1][2]);
    acc[1][3] = WMMA_F16(a1, b3, acc[1][3]);
  }

  const long long cb = (long long)blockIdx.z * c_bstride;
#pragma unroll
  for (int i = 0; i < 2; ++i) {
#pragma unroll
    for (int j = 0; j < 4; ++j) {
      const int ncol = n_blk + wn + j * 16 + ln;
#pragma unroll
      for (int r = 0; r < 8; ++r) {
        const int row = m_blk + wm + i * 16 + r + 8 * g;  // C layout: lane16+ -> M+8
        if (out_f16) {
          ((_Float16*)Cout)[cb + (long long)row * ldc + ncol] = (_Float16)acc[i][j][r];
        } else {
          ((float*)Cout)[cb + (long long)row * ldc + ncol] = acc[i][j][r];
        }
      }
    }
  }
}

// ---------------------------------------------------------------------------
// RoPE in-place on f16 tensor shaped (B, T, NH, HD). pos = t.
// one thread per (b, t, h, d<64) rotation pair
// ---------------------------------------------------------------------------
__global__ __launch_bounds__(256) void rope_f16(_Float16* __restrict__ x, int T) {
  long long idx = (long long)blockIdx.x * blockDim.x + threadIdx.x;
  const int d = (int)(idx & 63);
  long long r = idx >> 6;
  const int h = (int)(r & (NH - 1));
  r >>= 5;
  const int t = (int)(r % T);
  const int b = (int)(r / T);
  const float invf = __powf(10000.0f, -(float)d / 64.0f);
  float s, c;
  __sincosf((float)t * invf, &s, &c);
  _Float16* p = x + (((long long)(b * T + t)) * NH + h) * HD;
  const float x1 = (float)p[d];
  const float x2 = (float)p[d + 64];
  p[d]      = (_Float16)(x1 * c - x2 * s);
  p[d + 64] = (_Float16)(x2 * c + x1 * s);
}

// ---------------------------------------------------------------------------
// Fused attention. Grid: (SLOCAL/128, NH, BATCH), block = 256 (8 waves).
// Each wave: 16 queries, online softmax over all SEQ keys in 32-key blocks.
// V transposed into LDS cooperatively; P round-trips through per-wave LDS to
// convert from WMMA C-layout to A-fragment layout.
// ---------------------------------------------------------------------------
#define VT_STRIDE 48   // halves; 96B rows -> 16B-aligned b128 reads
#define P_STRIDE  40   // halves; 80B rows -> 16B-aligned b128 reads

__global__ __launch_bounds__(256) void attn_wmma(const _Float16* __restrict__ Q,
                                                 const _Float16* __restrict__ Kb,
                                                 const _Float16* __restrict__ Vb,
                                                 _Float16* __restrict__ O) {
  __shared__ __align__(16) _Float16 VtLds[HD * VT_STRIDE];        // [dim][key]
  __shared__ __align__(16) _Float16 Plds[8 * 16 * P_STRIDE];      // per-wave 16x32

  const int b = blockIdx.z;
  const int h = blockIdx.y;
  const int lane = threadIdx.x & 31;
  const int wave = threadIdx.x >> 5;
  const int ln = lane & 15;
  const int g  = lane >> 4;
  const int q0 = blockIdx.x * 128 + wave * 16;
  const float scale = 0.08838834764831845f;  // 1/sqrt(128)

  // Preload Q fragments (16 queries x 128 dims = 4 A-frags)
  const _Float16* qrow = Q + (((long long)(b * SLOCAL + q0 + ln)) * NH + h) * HD;
  v16h qa[4];
#pragma unroll
  for (int c = 0; c < 4; ++c) {
    v8h lo = *reinterpret_cast<const v8h*>(qrow + c * 32 + g * 16);
    v8h hi = *reinterpret_cast<const v8h*>(qrow + c * 32 + g * 16 + 8);
    qa[c] = cat8(lo, hi);
  }

  v8f zero = {0.f, 0.f, 0.f, 0.f, 0.f, 0.f, 0.f, 0.f};
  v8f oacc[8];
#pragma unroll
  for (int t = 0; t < 8; ++t) oacc[t] = zero;
  float mrow[8], lrow[8];
#pragma unroll
  for (int r = 0; r < 8; ++r) { mrow[r] = -INFINITY; lrow[r] = 0.f; }

  _Float16* myP = Plds + wave * 16 * P_STRIDE;

  const int vrow = threadIdx.x & 31;          // key within block
  const int vseg = (threadIdx.x >> 5) * 16;   // dim segment

  for (int j0 = 0; j0 < SEQ; j0 += 32) {
    __syncthreads();  // protect Vt overwrite vs prior iteration's reads
    // --- cooperative V transpose into LDS: Vt[dim][key] ---
    {
      const _Float16* vp =
          Vb + (((long long)(b * SEQ + j0 + vrow)) * NH + h) * HD + vseg;
      v8h x0 = *reinterpret_cast<const v8h*>(vp);
      v8h x1 = *reinterpret_cast<const v8h*>(vp + 8);
#pragma unroll
      for (int i = 0; i < 8; ++i) VtLds[(vseg + i) * VT_STRIDE + vrow] = x0[i];
#pragma unroll
      for (int i = 0; i < 8; ++i) VtLds[(vseg + 8 + i) * VT_STRIDE + vrow] = x1[i];
    }
    __syncthreads();

    // --- S = Q * K^T for 32 keys (two 16x16 score tiles) ---
    v8f s0 = zero, s1 = zero;
    const _Float16* krow0 =
        Kb + (((long long)(b * SEQ + j0 + ln)) * NH + h) * HD;
    const _Float16* krow1 = krow0 + (long long)16 * NH * HD;
#pragma unroll
    for (int c = 0; c < 4; ++c) {
      v16h kb0 = *reinterpret_cast<const v16h*>(krow0 + c * 32 + g * 16);
      v16h kb1 = *reinterpret_cast<const v16h*>(krow1 + c * 32 + g * 16);
      s0 = WMMA_F16(qa[c], kb0, s0);
      s1 = WMMA_F16(qa[c], kb1, s1);
    }

    // --- online softmax update (row r of this wave = query q0 + r + 8g) ---
    float alpha[8];
#pragma unroll
    for (int r = 0; r < 8; ++r) {
      s0[r] *= scale;
      s1[r] *= scale;
      float bm = redmax16(fmaxf(s0[r], s1[r]));
      float mn = fmaxf(mrow[r], bm);
      alpha[r] = __expf(mrow[r] - mn);
      mrow[r] = mn;
      s0[r] = __expf(s0[r] - mn);
      s1[r] = __expf(s1[r] - mn);
      lrow[r] = lrow[r] * alpha[r] + redsum16(s0[r] + s1[r]);
    }
#pragma unroll
    for (int t = 0; t < 8; ++t)
#pragma unroll
      for (int r = 0; r < 8; ++r) oacc[t][r] *= alpha[r];

    // --- P: C-layout f32 -> LDS -> A-fragment f16 ---
#pragma unroll
    for (int r = 0; r < 8; ++r) {
      const int row = r + 8 * g;
      myP[row * P_STRIDE + ln]      = (_Float16)s0[r];
      myP[row * P_STRIDE + 16 + ln] = (_Float16)s1[r];
    }
    asm volatile("s_wait_dscnt 0x0" ::: "memory");  // same-wave LDS RAW
    v16h pa;
    {
      const _Float16* pp = myP + ln * P_STRIDE;
      v8h lo = *reinterpret_cast<const v8h*>(pp + g * 16);
      v8h hi = *reinterpret_cast<const v8h*>(pp + g * 16 + 8);
      pa = cat8(lo, hi);
    }

    // --- O += P * V (8 dim-tiles) ---
#pragma unroll
    for (int t = 0; t < 8; ++t) {
      const _Float16* vtp = VtLds + (t * 16 + ln) * VT_STRIDE + g * 16;
      v16h vbf = *reinterpret_cast<const v16h*>(vtp);
      oacc[t] = WMMA_F16(pa, vbf, oacc[t]);
    }
  }

  // --- finalize: O /= l, store f16 (reference casts attn to fp16) ---
  float inv[8];
#pragma unroll
  for (int r = 0; r < 8; ++r) inv[r] = 1.0f / lrow[r];
#pragma unroll
  for (int t = 0; t < 8; ++t) {
#pragma unroll
    for (int r = 0; r < 8; ++r) {
      const int row = q0 + r + 8 * g;
      O[(((long long)(b * SLOCAL + row)) * NH + h) * HD + t * 16 + ln] =
          (_Float16)(oacc[t][r] * inv[r]);
    }
  }
}

// ---------------------------------------------------------------------------
// Host-side orchestration
// ---------------------------------------------------------------------------
extern "C" void kernel_launch(void* const* d_in, const int* in_sizes, int n_in,
                              void* d_out, int out_size, void* d_ws, size_t ws_size,
                              hipStream_t stream) {
  const float* hs = (const float*)d_in[0];
  const float* wq = (const float*)d_in[1];
  const float* wk = (const float*)d_in[2];
  const float* wv = (const float*)d_in[3];
  const float* wo = (const float*)d_in[4];
  float* out = (float*)d_out;

  const long long HS_N = (long long)BATCH * SEQ * HID;   // 33,554,432
  const long long W_N  = (long long)HID * HID;           // 16,777,216
  const long long Q_N  = (long long)BATCH * SLOCAL * HID;

  _Float16* ws = (_Float16*)d_ws;
  _Float16* hs16 = ws;                 long long o = HS_N;
  _Float16* wq16 = ws + o;             o += W_N;
  _Float16* wk16 = ws + o;             o += W_N;
  _Float16* wv16 = ws + o;             o += W_N;
  _Float16* wo16 = ws + o;             o += W_N;
  _Float16* q16  = ws + o;             o += Q_N;
  _Float16* k16  = ws + o;             o += HS_N;
  _Float16* v16  = ws + o;             o += HS_N;
  _Float16* at16 = ws + o;             o += Q_N;

  // 1) fp32 -> f16 conversions
  cvt_f32_f16<<<(unsigned)(HS_N / 4 / 256), 256, 0, stream>>>(hs, hs16, HS_N / 4);
  cvt_f32_f16<<<(unsigned)(W_N / 4 / 256), 256, 0, stream>>>(wq, wq16, W_N / 4);
  cvt_f32_f16<<<(unsigned)(W_N / 4 / 256), 256, 0, stream>>>(wk, wk16, W_N / 4);
  cvt_f32_f16<<<(unsigned)(W_N / 4 / 256), 256, 0, stream>>>(wv, wv16, W_N / 4);
  cvt_f32_f16<<<(unsigned)(W_N / 4 / 256), 256, 0, stream>>>(wo, wo16, W_N / 4);

  // 2) projections (all dims multiples of 128, no tails)
  // Q: per-batch M=1024 rows of hs
  gemm_wmma<<<dim3(HID / 128, SLOCAL / 128, BATCH), 256, 0, stream>>>(
      hs16, wq16, (void*)q16, HID, HID, HID, HID,
      (long long)SEQ * HID, (long long)SLOCAL * HID, 1);
  // K: M = B*SEQ contiguous
  gemm_wmma<<<dim3(HID / 128, (BATCH * SEQ) / 128, 1), 256, 0, stream>>>(
      hs16, wk16, (void*)k16, HID, HID, HID, HID, 0, 0, 1);
  // V
  gemm_wmma<<<dim3(HID / 128, (BATCH * SEQ) / 128, 1), 256, 0, stream>>>(
      hs16, wv16, (void*)v16, HID, HID, HID, HID, 0, 0, 1);

  // 3) RoPE (in place, f16)
  {
    long long nq = (long long)BATCH * SLOCAL * NH * 64;
    long long nk = (long long)BATCH * SEQ * NH * 64;
    rope_f16<<<(unsigned)(nq / 256), 256, 0, stream>>>(q16, SLOCAL);
    rope_f16<<<(unsigned)(nk / 256), 256, 0, stream>>>(k16, SEQ);
  }

  // 4) fused attention -> f16 attn
  attn_wmma<<<dim3(SLOCAL / 128, NH, BATCH), 256, 0, stream>>>(q16, k16, v16, at16);

  // 5) O-projection -> fp32 output (M = B*SLOCAL contiguous)
  gemm_wmma<<<dim3(HID / 128, (BATCH * SLOCAL) / 128, 1), 256, 0, stream>>>(
      at16, wo16, (void*)out, HID, HID, HID, HID, 0, 0, 0);

  (void)in_sizes; (void)n_in; (void)out_size; (void)ws_size;
}
